// Net_52587579572277
// MI455X (gfx1250) — matrix operand order repaired and checked by reading, not compiled, not run on hardware
//
#include <hip/hip_runtime.h>

#define SEQ 256
#define BATCH 64
#define HID 1024
#define G4 (4 * HID)   // 4096

typedef __attribute__((ext_vector_type(16))) __bf16 v16bf;
typedef __attribute__((ext_vector_type(8)))  __bf16 v8bf;
typedef __attribute__((ext_vector_type(8)))  float  v8f;

union AFrag { v16bf v; v8bf h[2]; };

__device__ __forceinline__ unsigned short f32_to_bf16(float f) {
    unsigned int u = __float_as_uint(f);
    u += 0x7FFFu + ((u >> 16) & 1u);   // round to nearest even
    return (unsigned short)(u >> 16);
}
__device__ __forceinline__ float bf16_to_f32(unsigned short h) {
    return __uint_as_float(((unsigned int)h) << 16);
}
__device__ __forceinline__ float sigmoidf(float x) {
    return 1.0f / (1.0f + __expf(-x));
}

// Split f32 array into bf16 hi + bf16 residual lo.
__global__ void k_split(const float* __restrict__ src,
                        unsigned short* __restrict__ hi,
                        unsigned short* __restrict__ lo, int n) {
    int i = blockIdx.x * 256 + threadIdx.x;
    if (i < n) {
        float f = src[i];
        unsigned short h = f32_to_bf16(f);
        hi[i] = h;
        lo[i] = f32_to_bf16(f - bf16_to_f32(h));
    }
}

// bias = b_ih + b_hh ; zero h (both ping-pong bf16 buffers), c, h_f32
__global__ void k_init(const float* __restrict__ b_ih, const float* __restrict__ b_hh,
                       float* __restrict__ bias,
                       unsigned short* __restrict__ h0hi, unsigned short* __restrict__ h0lo,
                       unsigned short* __restrict__ h1hi, unsigned short* __restrict__ h1lo,
                       float* __restrict__ c, float* __restrict__ hf) {
    int i = blockIdx.x * 256 + threadIdx.x;   // 65536 threads
    if (i < G4) bias[i] = b_ih[i] + b_hh[i];
    h0hi[i] = 0; h0lo[i] = 0; h1hi[i] = 0; h1lo[i] = 0;
    c[i] = 0.0f; hf[i] = 0.0f;
}

// x_proj[r][j] = sum_k x[r][k] * W_ih[j][k] + bias[j]   (r = s*B+b, 16384 rows)
// grid: (G4/16, 16384/128), block 256. Wave w owns rows [by*128+w*16, +16), cols [bx*16, +16).
__global__ __launch_bounds__(256)
void k_xproj(const unsigned short* __restrict__ xhi_, const unsigned short* __restrict__ xlo_,
             const unsigned short* __restrict__ whi_, const unsigned short* __restrict__ wlo_,
             const float* __restrict__ bias, float* __restrict__ xproj) {
    const __bf16* xhi = reinterpret_cast<const __bf16*>(xhi_);
    const __bf16* xlo = reinterpret_cast<const __bf16*>(xlo_);
    const __bf16* whi = reinterpret_cast<const __bf16*>(whi_);
    const __bf16* wlo = reinterpret_cast<const __bf16*>(wlo_);

    const int tid = threadIdx.x, wave = tid >> 5, lane = tid & 31;
    const int n = lane & 15, khalf = lane >> 4;
    const int j0 = blockIdx.x * 16;
    const int mrow0 = blockIdx.y * 128 + wave * 16;
    const int r = mrow0 + n;                    // A row for this lane

    const __bf16* Ah = xhi + (size_t)r * HID;
    const __bf16* Al = xlo + (size_t)r * HID;
    const __bf16* Bh = whi + (size_t)(j0 + n) * HID;
    const __bf16* Bl = wlo + (size_t)(j0 + n) * HID;

    v8f acc = {};    // hi*hi chain
    v8f acc2 = {};   // cross-term chain (independent -> 2x WMMA-pipe ILP)
    for (int kb = 0; kb < HID; kb += 32) {
        AFrag ah, al;
        ah.h[0] = *(const v8bf*)(Ah + kb + khalf * 16);
        ah.h[1] = *(const v8bf*)(Ah + kb + 8 + khalf * 16);
        al.h[0] = *(const v8bf*)(Al + kb + khalf * 16);
        al.h[1] = *(const v8bf*)(Al + kb + 8 + khalf * 16);
        v16bf bh = *(const v16bf*)(Bh + kb + khalf * 16);
        v16bf bl = *(const v16bf*)(Bl + kb + khalf * 16);
        acc  = __builtin_amdgcn_wmma_f32_16x16x32_bf16(false, ah.v, false, bh, (short)0, acc,  false, false);
        acc2 = __builtin_amdgcn_wmma_f32_16x16x32_bf16(false, ah.v, false, bl, (short)0, acc2, false, false);
        acc2 = __builtin_amdgcn_wmma_f32_16x16x32_bf16(false, al.v, false, bh, (short)0, acc2, false, false);
    }
    const float bn = bias[j0 + n];
#pragma unroll
    for (int v = 0; v < 8; ++v) {
        int row = mrow0 + v + khalf * 8;
        xproj[(size_t)row * G4 + j0 + n] = acc[v] + acc2[v] + bn;
    }
}

// One recurrent step: gates = x_proj[t] + h_prev @ W_hh^T, then fused LSTM pointwise.
// grid 128, block 512 (16 waves).
//   blockIdx.x&1 -> batch block of 32, blockIdx.x>>1 -> hidden-unit block of 16.
//   wave: gate = w&3, mt = (w>>2)&1, ksec = w>>3 (K split in halves of 512).
__global__ __launch_bounds__(512)
void k_step(int t, const float* __restrict__ xproj,
            const unsigned short* __restrict__ whi_, const unsigned short* __restrict__ wlo_,
            const unsigned short* __restrict__ hphi_, const unsigned short* __restrict__ hplo_,
            unsigned short* __restrict__ hnhi, unsigned short* __restrict__ hnlo,
            float* __restrict__ c, float* __restrict__ hf) {
    const __bf16* whi = reinterpret_cast<const __bf16*>(whi_);
    const __bf16* wlo = reinterpret_cast<const __bf16*>(wlo_);
    const __bf16* hphi = reinterpret_cast<const __bf16*>(hphi_);
    const __bf16* hplo = reinterpret_cast<const __bf16*>(hplo_);

    const int tid = threadIdx.x, wave = tid >> 5, lane = tid & 31;
    const int n = lane & 15, khalf = lane >> 4;
    const int m0 = (blockIdx.x & 1) * 32;        // batch block origin
    const int u0 = (blockIdx.x >> 1) * 16;       // hidden-unit block origin
    const int gate = wave & 3;                   // 0=i 1=f 2=g 3=o
    const int mt = (wave >> 2) & 1;              // batch tile within block
    const int ksec = wave >> 3;                  // K section (0/1)
    const int j0 = gate * HID + u0;              // gate column tile origin
    const int batch = m0 + mt * 16 + n;          // A row for this lane
    const int k0 = ksec * (HID / 2);

    const __bf16* Ah = hphi + (size_t)batch * HID + k0;
    const __bf16* Al = hplo + (size_t)batch * HID + k0;
    const __bf16* Bh = whi + (size_t)(j0 + n) * HID + k0;
    const __bf16* Bl = wlo + (size_t)(j0 + n) * HID + k0;

    v8f acc = {};    // hi*hi chain
    v8f acc2 = {};   // cross-term chain
    for (int kb = 0; kb < HID / 2; kb += 32) {
        AFrag ah, al;
        ah.h[0] = *(const v8bf*)(Ah + kb + khalf * 16);
        ah.h[1] = *(const v8bf*)(Ah + kb + 8 + khalf * 16);
        al.h[0] = *(const v8bf*)(Al + kb + khalf * 16);
        al.h[1] = *(const v8bf*)(Al + kb + 8 + khalf * 16);
        v16bf bh = *(const v16bf*)(Bh + kb + khalf * 16);
        v16bf bl = *(const v16bf*)(Bl + kb + khalf * 16);
        acc  = __builtin_amdgcn_wmma_f32_16x16x32_bf16(false, ah.v, false, bh, (short)0, acc,  false, false);
        acc2 = __builtin_amdgcn_wmma_f32_16x16x32_bf16(false, ah.v, false, bl, (short)0, acc2, false, false);
        acc2 = __builtin_amdgcn_wmma_f32_16x16x32_bf16(false, al.v, false, bh, (short)0, acc2, false, false);
    }

    __shared__ float g_lds[4][2][32][17];   // [gate][ksec][batch-in-block][unit], padded
#pragma unroll
    for (int v = 0; v < 8; ++v)
        g_lds[gate][ksec][mt * 16 + v + khalf * 8][n] = acc[v] + acc2[v];
    __syncthreads();

    // fused pointwise LSTM update: 32x16 = 512 elements, 1 per thread
    {
        int ml = tid >> 4, ul = tid & 15;
        int b = m0 + ml, u = u0 + ul;
        const float* xp = xproj + (size_t)(t * BATCH + b) * G4;
        float gi = g_lds[0][0][ml][ul] + g_lds[0][1][ml][ul] + xp[u];
        float gf = g_lds[1][0][ml][ul] + g_lds[1][1][ml][ul] + xp[HID + u];
        float gg = g_lds[2][0][ml][ul] + g_lds[2][1][ml][ul] + xp[2 * HID + u];
        float go = g_lds[3][0][ml][ul] + g_lds[3][1][ml][ul] + xp[3 * HID + u];
        size_t idx = (size_t)b * HID + u;
        float cv = sigmoidf(gf) * c[idx] + sigmoidf(gi) * tanhf(gg);
        float hv = sigmoidf(go) * tanhf(cv);
        c[idx] = cv;
        hf[idx] = hv;
        unsigned short hh = f32_to_bf16(hv);
        hnhi[idx] = hh;
        hnlo[idx] = f32_to_bf16(hv - bf16_to_f32(hh));
    }
}

extern "C" void kernel_launch(void* const* d_in, const int* in_sizes, int n_in,
                              void* d_out, int out_size, void* d_ws, size_t ws_size,
                              hipStream_t stream) {
    const float* x    = (const float*)d_in[0];   // [S,B,H]
    const float* W_ih = (const float*)d_in[1];   // [4H,H]
    const float* W_hh = (const float*)d_in[2];   // [4H,H]
    const float* b_ih = (const float*)d_in[3];   // [4H]
    const float* b_hh = (const float*)d_in[4];   // [4H]

    char* p = (char*)d_ws;
    auto alloc = [&](size_t bytes) -> void* {
        void* r = (void*)p;
        p += (bytes + 255) & ~(size_t)255;
        return r;
    };
    const size_t NX  = (size_t)SEQ * BATCH * HID;   // 16,777,216
    const size_t NW  = (size_t)G4 * HID;            // 4,194,304
    const size_t NH  = (size_t)BATCH * HID;         // 65,536

    float*          xproj  = (float*)alloc(NX * 4ull * 4);  // [S*B, 4H] f32
    unsigned short* whh_hi = (unsigned short*)alloc(NW * 2);
    unsigned short* whh_lo = (unsigned short*)alloc(NW * 2);
    unsigned short* wih_hi = (unsigned short*)alloc(NW * 2);
    unsigned short* wih_lo = (unsigned short*)alloc(NW * 2);
    unsigned short* x_hi   = (unsigned short*)alloc(NX * 2);
    unsigned short* x_lo   = (unsigned short*)alloc(NX * 2);
    unsigned short* h_hi0  = (unsigned short*)alloc(NH * 2);
    unsigned short* h_lo0  = (unsigned short*)alloc(NH * 2);
    unsigned short* h_hi1  = (unsigned short*)alloc(NH * 2);
    unsigned short* h_lo1  = (unsigned short*)alloc(NH * 2);
    float*          c_buf  = (float*)alloc(NH * 4);
    float*          h_f32  = (float*)alloc(NH * 4);
    float*          bias   = (float*)alloc((size_t)G4 * 4);

    // Precision splits + init (recomputed every call -> deterministic)
    k_split<<<(int)((NW + 255) / 256), 256, 0, stream>>>(W_ih, wih_hi, wih_lo, (int)NW);
    k_split<<<(int)((NW + 255) / 256), 256, 0, stream>>>(W_hh, whh_hi, whh_lo, (int)NW);
    k_split<<<(int)((NX + 255) / 256), 256, 0, stream>>>(x, x_hi, x_lo, (int)NX);
    k_init<<<256, 256, 0, stream>>>(b_ih, b_hh, bias, h_hi0, h_lo0, h_hi1, h_lo1, c_buf, h_f32);

    // Big parallel pre-projection GEMM
    k_xproj<<<dim3(G4 / 16, (SEQ * BATCH) / 128), 256, 0, stream>>>(
        x_hi, x_lo, wih_hi, wih_lo, bias, xproj);

    // Serial recurrence, ping-pong h buffers
    for (int t = 0; t < SEQ; ++t) {
        const unsigned short* rhi = (t & 1) ? h_hi1 : h_hi0;
        const unsigned short* rlo = (t & 1) ? h_lo1 : h_lo0;
        unsigned short* whi = (t & 1) ? h_hi0 : h_hi1;
        unsigned short* wlo = (t & 1) ? h_lo0 : h_lo1;
        k_step<<<128, 512, 0, stream>>>(t, xproj, whh_hi, whh_lo,
                                        rhi, rlo, whi, wlo, c_buf, h_f32);
    }

    hipMemcpyAsync(d_out, h_f32, NH * sizeof(float), hipMemcpyDeviceToDevice, stream);
}
